// AttnBlock_24404004176504
// MI455X (gfx1250) — matrix operand order repaired and checked by reading, not compiled, not run on hardware
//
#include <hip/hip_runtime.h>

// ---------------------------------------------------------------------------
// AttnBlock for (b=2, C=256, 16x16x16) on gfx1250 (MI455X), bf16 WMMA path.
// ---------------------------------------------------------------------------

#define Cc   256
#define Nn   4096
#define Bb   2
#define Gg   32
#define CG   8          // channels per group
#define SCALE 0.0625f   // 1/sqrt(256)

typedef __attribute__((ext_vector_type(16))) __bf16        v16bf;
typedef __attribute__((ext_vector_type(8)))  float         v8f;
typedef __attribute__((ext_vector_type(4)))  unsigned int  v4u;

union Frag {
    v16bf v;
    unsigned short u[16];
    v4u q[2];           // q[0] = halves 0..7, q[1] = halves 8..15
};

__device__ __forceinline__ unsigned short f2bf(float f) {
    unsigned int u = __float_as_uint(f);
    unsigned int r = u + 0x7FFFu + ((u >> 16) & 1u);   // round-to-nearest-even
    return (unsigned short)(r >> 16);
}

__device__ __forceinline__ v8f wmma_bf16(v16bf a, v16bf b, v8f c) {
    // (neg_a, A, neg_b, B, c_mod, C, reuse_a, reuse_b)
    return __builtin_amdgcn_wmma_f32_16x16x32_bf16(false, a, false, b,
                                                   (short)0, c, false, false);
}

// Pack 8 f32 -> 8 bf16 and store as one 16B chunk.
__device__ __forceinline__ void store8bf(unsigned short* p, const float* f) {
    union { v4u q; unsigned short u[8]; } c;
#pragma unroll
    for (int i = 0; i < 8; ++i) c.u[i] = f2bf(f[i]);
    *(v4u*)p = c.q;
}

// ---------------------------------------------------------------------------
// Kernel 0: convert the four 256x256 f32 weight matrices to bf16.
// ---------------------------------------------------------------------------
__global__ __launch_bounds__(256) void cvtw_kernel(
    const float* __restrict__ wq, const float* __restrict__ wk,
    const float* __restrict__ wv, const float* __restrict__ wp,
    unsigned short* __restrict__ out)
{
    int which = blockIdx.y;
    const float* src = (which == 0) ? wq : (which == 1) ? wk : (which == 2) ? wv : wp;
    int i = blockIdx.x * 256 + threadIdx.x;
    out[(size_t)which * (Cc * Cc) + i] = f2bf(src[i]);
}

// ---------------------------------------------------------------------------
// Kernel 1: GroupNorm -> bf16, channel-last layout hn[b][n][c].
// One block per (group, batch): reduce 8*4096 elements.
// ---------------------------------------------------------------------------
__global__ __launch_bounds__(256) void gn_kernel(
    const float* __restrict__ x, const float* __restrict__ gamma,
    const float* __restrict__ beta, unsigned short* __restrict__ hn_nc)
{
    const int g = blockIdx.x, b = blockIdx.y, tid = threadIdx.x;
    const int M = CG * Nn;                           // 32768
    const float* xb = x + ((size_t)b * Cc + g * CG) * (size_t)Nn;

    float s = 0.f, ss = 0.f;
    for (int i = tid; i < M; i += 256) { float v = xb[i]; s += v; ss += v * v; }

    __shared__ float rs[256], rq[256];
    rs[tid] = s; rq[tid] = ss; __syncthreads();
#pragma unroll
    for (int o = 128; o > 0; o >>= 1) {
        if (tid < o) { rs[tid] += rs[tid + o]; rq[tid] += rq[tid + o]; }
        __syncthreads();
    }
    const float mean = rs[0] * (1.f / (float)M);
    const float var  = rq[0] * (1.f / (float)M) - mean * mean;
    const float inv  = rsqrtf(var + 1e-6f);

    for (int i = tid; i < M; i += 256) {
        int j = i >> 12;            // channel within group
        int n = i & (Nn - 1);
        int c = g * CG + j;
        float v = (xb[i] - mean) * inv * gamma[c] + beta[c];
        hn_nc[((size_t)b * Nn + n) * Cc + c] = f2bf(v);
    }
}

// ---------------------------------------------------------------------------
// Kernel 2: QKV projection. out[o,n] = W[o,:] . hn[:,n] + bias[o]
// grid (N/64, 3, B), 8 waves; each wave: 2 o-tiles x 4 n-tiles, K-loop by 32.
// q -> qT[b][n][c], k -> kT[b][m][c], v -> v_cn[b][c][m]
// ---------------------------------------------------------------------------
__global__ __launch_bounds__(256) void qkv_kernel(
    const unsigned short* __restrict__ wbf,   // 3 x 256x256 bf16
    const float* __restrict__ bq, const float* __restrict__ bk,
    const float* __restrict__ bv,
    const unsigned short* __restrict__ hn_nc,
    unsigned short* __restrict__ qT, unsigned short* __restrict__ kT,
    unsigned short* __restrict__ v_cn)
{
    const int nb = blockIdx.x, which = blockIdx.y, b = blockIdx.z;
    const int tid = threadIdx.x, w = tid >> 5, lane = tid & 31;
    const int hi = lane >> 4, l15 = lane & 15;
    const unsigned short* W = wbf + (size_t)which * (Cc * Cc);
    const float* bias = (which == 0) ? bq : (which == 1) ? bk : bv;

    __shared__ float sbias[Cc];
    sbias[tid] = bias[tid];
    __syncthreads();

    const int o0 = w * 32;
    const int n0 = nb * 64;

    v8f acc[2][4];
#pragma unroll
    for (int ot = 0; ot < 2; ++ot)
#pragma unroll
        for (int nt = 0; nt < 4; ++nt) acc[ot][nt] = (v8f){};

    for (int cb = 0; cb < Cc; cb += 32) {
        Frag a[2];
#pragma unroll
        for (int ot = 0; ot < 2; ++ot) {
            const unsigned short* rp = W + (size_t)(o0 + ot * 16 + l15) * Cc + cb;
            a[ot].q[0] = *(const v4u*)(rp + hi * 8);
            a[ot].q[1] = *(const v4u*)(rp + 16 + hi * 8);
        }
        Frag bf[4];
#pragma unroll
        for (int nt = 0; nt < 4; ++nt) {
            const unsigned short* bp =
                hn_nc + ((size_t)b * Nn + n0 + nt * 16 + l15) * Cc + cb + hi * 8;
            bf[nt].q[0] = *(const v4u*)bp;
            bf[nt].q[1] = *(const v4u*)(bp + 16);
            if (cb + 32 < Cc) __builtin_prefetch(bp + 32, 0, 3);
        }
#pragma unroll
        for (int ot = 0; ot < 2; ++ot)
#pragma unroll
            for (int nt = 0; nt < 4; ++nt)
                acc[ot][nt] = wmma_bf16(a[ot].v, bf[nt].v, acc[ot][nt]);
    }

#pragma unroll
    for (int ot = 0; ot < 2; ++ot) {
        const int ob = o0 + ot * 16 + 8 * hi;           // first of 8 contiguous o
#pragma unroll
        for (int nt = 0; nt < 4; ++nt) {
            const int ncol = n0 + nt * 16 + l15;
            float vals[8];
#pragma unroll
            for (int r = 0; r < 8; ++r) vals[r] = acc[ot][nt][r] + sbias[ob + r];
            if (which == 0) {
                store8bf(qT + ((size_t)b * Nn + ncol) * Cc + ob, vals);
            } else if (which == 1) {
                store8bf(kT + ((size_t)b * Nn + ncol) * Cc + ob, vals);
            } else {
#pragma unroll
                for (int r = 0; r < 8; ++r)
                    v_cn[((size_t)b * Cc + ob + r) * Nn + ncol] = f2bf(vals[r]);
            }
        }
    }
}

// ---------------------------------------------------------------------------
// Kernel 3: fused flash attention, 64 keys per iteration.
// grid (N/16, B); block = 256 = 8 waves. Each block: 16 query rows, all 256 c.
// Wave w: K-slice c[32w..32w+32) of Q^T K (summed via LDS, deterministic),
//         owns output channels c[32w..32w+32) of O = P V^T, and owns query
//         rows {2w, 2w+1} for the parallel online-softmax update.
// ---------------------------------------------------------------------------
__global__ __launch_bounds__(256) void attn_kernel(
    const unsigned short* __restrict__ qT, const unsigned short* __restrict__ kT,
    const unsigned short* __restrict__ v_cn, unsigned short* __restrict__ h_nc)
{
    const int nb = blockIdx.x, b = blockIdx.y;
    const int tid = threadIdx.x, w = tid >> 5, lane = tid & 31;
    const int hi = lane >> 4, l15 = lane & 15;
    const int n0 = nb * 16;
    const int cb = w * 32;

    __shared__ float sPart[8][16][64];                       // 32 KB S partials
    __shared__ __align__(16) unsigned short sP[16][64];      // bf16 P tile
    __shared__ float sAlpha[16], sM[16], sL[16];

    if (tid < 16) { sM[tid] = -3.0e38f; sL[tid] = 0.f; }

    // Q A-fragment for this wave's c-slice (kept in registers for all steps)
    Frag qa;
    {
        const unsigned short* qrp = qT + ((size_t)b * Nn + n0 + l15) * Cc + cb;
        qa.q[0] = *(const v4u*)(qrp + hi * 8);
        qa.q[1] = *(const v4u*)(qrp + 16 + hi * 8);
    }
    v8f oAcc0 = (v8f){}, oAcc1 = (v8f){};
    __syncthreads();

    for (int m0 = 0; m0 < Nn; m0 += 64) {
        // --- partial S = (q slice)^T (k slice) for 64 keys (4 m-tiles) ---
        Frag kb[4];
#pragma unroll
        for (int mt = 0; mt < 4; ++mt) {
            const unsigned short* kp =
                kT + ((size_t)b * Nn + m0 + mt * 16 + l15) * Cc + cb + hi * 8;
            kb[mt].q[0] = *(const v4u*)kp;
            kb[mt].q[1] = *(const v4u*)(kp + 16);
        }
        const v8f z = (v8f){};
#pragma unroll
        for (int mt = 0; mt < 4; ++mt) {
            v8f s = wmma_bf16(qa.v, kb[mt].v, z);
#pragma unroll
            for (int r = 0; r < 8; ++r)
                sPart[w][r + 8 * hi][mt * 16 + l15] = s[r];
        }
        __syncthreads();

        // --- parallel deterministic cross-wave sum + online softmax ---
        // wave w owns rows 2w (lanes 0-15) and 2w+1 (lanes 16-31);
        // each lane covers 4 key columns: l15, l15+16, l15+32, l15+48.
        {
            const int row = 2 * w + hi;
            float sv[4];
            float mx = -3.0e38f;
#pragma unroll
            for (int j = 0; j < 4; ++j) {
                float a = 0.f;
#pragma unroll
                for (int ww = 0; ww < 8; ++ww) a += sPart[ww][row][l15 + 16 * j];
                a *= SCALE;
                sv[j] = a;
                mx = fmaxf(mx, a);
            }
            // butterfly reductions within each 16-lane half (rows never mix)
#pragma unroll
            for (int d = 1; d < 16; d <<= 1) mx = fmaxf(mx, __shfl_xor(mx, d, 32));

            const float mOld = sM[row];
            const float mNew = fmaxf(mOld, mx);
            const float alpha = __expf(mOld - mNew);
            float ls = 0.f;
#pragma unroll
            for (int j = 0; j < 4; ++j) {
                float p = __expf(sv[j] - mNew);
                ls += p;
                sP[row][l15 + 16 * j] = f2bf(p);
            }
#pragma unroll
            for (int d = 1; d < 16; d <<= 1) ls += __shfl_xor(ls, d, 32);

            if (l15 == 0) {
                sM[row] = mNew;
                sL[row] = sL[row] * alpha + ls;
                sAlpha[row] = alpha;
            }
        }
        __syncthreads();

        // --- rescale running O, then O += P x V^T (K = 64 in two chunks) ---
        Frag pa0, pa1;
        {
            const unsigned short* pp = &sP[l15][0];
            pa0.q[0] = *(const v4u*)(pp + hi * 8);
            pa0.q[1] = *(const v4u*)(pp + 16 + hi * 8);
            pa1.q[0] = *(const v4u*)(pp + 32 + hi * 8);
            pa1.q[1] = *(const v4u*)(pp + 48 + hi * 8);
        }
        Frag vb[2][2];
#pragma unroll
        for (int ct = 0; ct < 2; ++ct) {
            const unsigned short* vp =
                v_cn + ((size_t)b * Cc + cb + ct * 16 + l15) * Nn + m0 + hi * 8;
            vb[ct][0].q[0] = *(const v4u*)vp;
            vb[ct][0].q[1] = *(const v4u*)(vp + 16);
            vb[ct][1].q[0] = *(const v4u*)(vp + 32);
            vb[ct][1].q[1] = *(const v4u*)(vp + 48);
        }
#pragma unroll
        for (int r = 0; r < 8; ++r) {
            const float al = sAlpha[r + 8 * hi];
            oAcc0[r] *= al;
            oAcc1[r] *= al;
        }
        oAcc0 = wmma_bf16(pa0.v, vb[0][0].v, oAcc0);
        oAcc0 = wmma_bf16(pa1.v, vb[0][1].v, oAcc0);
        oAcc1 = wmma_bf16(pa0.v, vb[1][0].v, oAcc1);
        oAcc1 = wmma_bf16(pa1.v, vb[1][1].v, oAcc1);
        __syncthreads();
    }

    // --- finalize: divide by softmax denominator, store h_[b][n][c] bf16 ---
#pragma unroll
    for (int r = 0; r < 8; ++r) {
        const int row = r + 8 * hi;
        const float inv = 1.f / sL[row];
        const size_t base = ((size_t)b * Nn + n0 + row) * Cc + cb + l15;
        h_nc[base]      = f2bf(oAcc0[r] * inv);
        h_nc[base + 16] = f2bf(oAcc1[r] * inv);
    }
}

// ---------------------------------------------------------------------------
// Kernel 4: output projection + bias + residual (fp32 out).
// out[b,o,n] = x[b,o,n] + wp[o,:] . h_[:,n] + bp[o]
// ---------------------------------------------------------------------------
__global__ __launch_bounds__(256) void proj_kernel(
    const unsigned short* __restrict__ wpbf, const float* __restrict__ bp,
    const unsigned short* __restrict__ h_nc, const float* __restrict__ x,
    float* __restrict__ out)
{
    const int nb = blockIdx.x, b = blockIdx.y;
    const int tid = threadIdx.x, w = tid >> 5, lane = tid & 31;
    const int hi = lane >> 4, l15 = lane & 15;

    __shared__ float sbias[Cc];
    sbias[tid] = bp[tid];
    __syncthreads();

    const int o0 = w * 32;
    const int n0 = nb * 64;

    v8f acc[2][4];
#pragma unroll
    for (int ot = 0; ot < 2; ++ot)
#pragma unroll
        for (int nt = 0; nt < 4; ++nt) acc[ot][nt] = (v8f){};

    for (int cb = 0; cb < Cc; cb += 32) {
        Frag a[2];
#pragma unroll
        for (int ot = 0; ot < 2; ++ot) {
            const unsigned short* rp = wpbf + (size_t)(o0 + ot * 16 + l15) * Cc + cb;
            a[ot].q[0] = *(const v4u*)(rp + hi * 8);
            a[ot].q[1] = *(const v4u*)(rp + 16 + hi * 8);
        }
        Frag bf[4];
#pragma unroll
        for (int nt = 0; nt < 4; ++nt) {
            const unsigned short* bpr =
                h_nc + ((size_t)b * Nn + n0 + nt * 16 + l15) * Cc + cb + hi * 8;
            bf[nt].q[0] = *(const v4u*)bpr;
            bf[nt].q[1] = *(const v4u*)(bpr + 16);
            if (cb + 32 < Cc) __builtin_prefetch(bpr + 32, 0, 3);
        }
#pragma unroll
        for (int ot = 0; ot < 2; ++ot)
#pragma unroll
            for (int nt = 0; nt < 4; ++nt)
                acc[ot][nt] = wmma_bf16(a[ot].v, bf[nt].v, acc[ot][nt]);
    }

#pragma unroll
    for (int ot = 0; ot < 2; ++ot) {
        const int ob = o0 + ot * 16 + 8 * hi;
#pragma unroll
        for (int nt = 0; nt < 4; ++nt) {
            const int ncol = n0 + nt * 16 + l15;
#pragma unroll
            for (int r = 0; r < 8; ++r) {
                const size_t idx = ((size_t)b * Cc + ob + r) * Nn + ncol;
                out[idx] = x[idx] + acc[ot][nt][r] + sbias[ob + r];
            }
        }
    }
}

// ---------------------------------------------------------------------------
// Host-side launcher.
// ---------------------------------------------------------------------------
extern "C" void kernel_launch(void* const* d_in, const int* in_sizes, int n_in,
                              void* d_out, int out_size, void* d_ws, size_t ws_size,
                              hipStream_t stream)
{
    (void)in_sizes; (void)n_in; (void)out_size; (void)ws_size;

    const float* x     = (const float*)d_in[0];
    const float* gamma = (const float*)d_in[1];
    const float* beta  = (const float*)d_in[2];
    const float* wq    = (const float*)d_in[3];
    const float* bq    = (const float*)d_in[4];
    const float* wk    = (const float*)d_in[5];
    const float* bk    = (const float*)d_in[6];
    const float* wv    = (const float*)d_in[7];
    const float* bv    = (const float*)d_in[8];
    const float* wp    = (const float*)d_in[9];
    const float* bp    = (const float*)d_in[10];
    float* out = (float*)d_out;

    char* ws = (char*)d_ws;
    const size_t WSZ  = (size_t)Cc * Cc * sizeof(unsigned short);      // 128 KB
    const size_t ASZ  = (size_t)Bb * Nn * Cc * sizeof(unsigned short); // 4 MB
    unsigned short* wbf  = (unsigned short*)ws;                        // 4 weights
    unsigned short* hn   = (unsigned short*)(ws + 4 * WSZ);
    unsigned short* qT   = (unsigned short*)(ws + 4 * WSZ + 1 * ASZ);
    unsigned short* kT   = (unsigned short*)(ws + 4 * WSZ + 2 * ASZ);
    unsigned short* vcn  = (unsigned short*)(ws + 4 * WSZ + 3 * ASZ);
    unsigned short* hnc  = (unsigned short*)(ws + 4 * WSZ + 4 * ASZ);

    cvtw_kernel<<<dim3(Cc * Cc / 256, 4), 256, 0, stream>>>(wq, wk, wv, wp, wbf);
    gn_kernel  <<<dim3(Gg, Bb),          256, 0, stream>>>(x, gamma, beta, hn);
    qkv_kernel <<<dim3(Nn / 64, 3, Bb),  256, 0, stream>>>(wbf, bq, bk, bv, hn,
                                                           qT, kT, vcn);
    attn_kernel<<<dim3(Nn / 16, Bb),     256, 0, stream>>>(qT, kT, vcn, hnc);
    proj_kernel<<<dim3(Nn / 64, Bb),     256, 0, stream>>>(
        wbf + 3 * (size_t)Cc * Cc, bp, hnc, x, out);
}
